// LocalSpatialAttention_2946347565627
// MI455X (gfx1250) — compile-verified, matrix operand order checked by reading, not run on hardware
//
#include <hip/hip_runtime.h>
#include <hip/hip_bf16.h>

// ---------------------------------------------------------------------------
// MI455X (gfx1250) implementation.
//
// Roofline: ~160 MB mandatory HBM traffic (idx 65MB + mask 16MB + x 38MB +
// out 38MB) => ~7us floor @ 23.3 TB/s. The 2.07 GFLOP einsum runs on
// v_wmma_f32_16x16x32_f16: per branch the contraction is G[N,216] x Wd[216,8];
// pad 216->224 = 7x32 K-chunks => 7 WMMAs/branch + 2 for att = 30 WMMAs per
// 16-voxel tile. att (2.4MB f16) is L2-resident for the 16.2M gathers, so
// gathers are issued UNCONDITIONALLY and zeroed with cndmask (no exec-mask
// branches); idx/mask stream past L2 with non-temporal hints.
// ---------------------------------------------------------------------------

typedef __attribute__((ext_vector_type(16))) _Float16 v16h;
typedef __attribute__((ext_vector_type(8)))  _Float16 v8h;
typedef __attribute__((ext_vector_type(8)))  float    v8f;
typedef __attribute__((ext_vector_type(2)))  float    v2f;

#define C_CH 64
#define R_CH 8
#define K_OFF 27
#define B_BR 4
#define KBLKS 7   // ceil(27/4): 4 kernel-offsets (4*8=32 K-dims) per WMMA

// ---------------------------------------------------------------------------
// Prep: repack W1 (64x8) and Wd (B,27,8,8) into f16 WMMA B-matrix layouts.
// 16-bit B 32x16 layout: lane L holds column s=L&15; kb=(L>>4)*8;
// elements 0..7 = K-dims kb..kb+7, elements 8..15 = K-dims kb+16..kb+23.
// Columns s>=8 and kernel-offsets k>=27 are zero-padded.
// ---------------------------------------------------------------------------
__global__ void lsa_prep(const float* __restrict__ W1, const float* __restrict__ Wd,
                         _Float16* __restrict__ W1B, _Float16* __restrict__ WdB) {
    int t = blockIdx.x * blockDim.x + threadIdx.x;
    if (t < B_BR * KBLKS * 32) {
        // WdB[(b*7+kblk)*32 + L][16]
        int b = t / (KBLKS * 32);
        int rem = t % (KBLKS * 32);
        int kblk = rem / 32;
        int L = rem % 32;
        int s = L & 15;
        int hk = L >> 4;
        int k_lo = 4 * kblk + hk;       // K-dim = k*8 + r ; k = kd>>3, r = kd&7
        int k_hi = k_lo + 2;
        _Float16* dst = WdB + (size_t)t * 16;
#pragma unroll
        for (int i = 0; i < 8; ++i) {
            float vlo = (s < R_CH && k_lo < K_OFF)
                ? Wd[(((size_t)b * K_OFF + k_lo) * R_CH + i) * R_CH + s] : 0.0f;
            float vhi = (s < R_CH && k_hi < K_OFF)
                ? Wd[(((size_t)b * K_OFF + k_hi) * R_CH + i) * R_CH + s] : 0.0f;
            dst[i]     = (_Float16)vlo;
            dst[8 + i] = (_Float16)vhi;
        }
    } else if (t < B_BR * KBLKS * 32 + 64) {
        // W1B[kblk2*32 + L][16], kblk2 selects channels 0..31 / 32..63
        int u = t - B_BR * KBLKS * 32;
        int kblk2 = u >> 5;
        int L = u & 31;
        int s = L & 15;
        int kb = (L >> 4) * 8;
        int base = kblk2 * 32;
        _Float16* dst = W1B + (size_t)u * 16;
#pragma unroll
        for (int i = 0; i < 8; ++i) {
            int rlo = base + kb + i;
            int rhi = base + kb + 16 + i;
            dst[i]     = (_Float16)((s < R_CH) ? W1[rlo * R_CH + s] : 0.0f);
            dst[8 + i] = (_Float16)((s < R_CH) ? W1[rhi * R_CH + s] : 0.0f);
        }
    }
}

__device__ __forceinline__ void pack4(v16h& a, int base, float4 f) {
    a[base + 0] = (_Float16)f.x; a[base + 1] = (_Float16)f.y;
    a[base + 2] = (_Float16)f.z; a[base + 3] = (_Float16)f.w;
}

// ---------------------------------------------------------------------------
// att = relu(x @ W1 + b1), stored as f16 rows of 8 (16B, L2-resident for the
// gather stage). One wave per 16-voxel tile; 2 WMMAs (K=64 in two 32-chunks).
// ---------------------------------------------------------------------------
__global__ void __launch_bounds__(256)
lsa_att(const float* __restrict__ x, const float* __restrict__ b1,
        const _Float16* __restrict__ W1B,
        _Float16* __restrict__ att_h, int Nv) {
    int lane = threadIdx.x & 31;
    int wave = blockIdx.x * (blockDim.x >> 5) + (threadIdx.x >> 5);
    int tiles = (Nv + 15) >> 4;
    if (wave >= tiles) return;
    int n0 = wave << 4;
    int s  = lane & 15;
    int hk = lane >> 4;

    int row = n0 + s;                 // A-matrix row for this lane (M = lane&15)
    if (row >= Nv) row = Nv - 1;      // clamp (only possible in a partial tail tile)
    const float* xr = x + (size_t)row * C_CH;
    int kb = hk * 8;

    // A tile for K-dims 0..31 (16-bit A 16x32 layout)
    v16h a0, a1;
    pack4(a0, 0,  *(const float4*)(xr + kb));
    pack4(a0, 4,  *(const float4*)(xr + kb + 4));
    pack4(a0, 8,  *(const float4*)(xr + kb + 16));
    pack4(a0, 12, *(const float4*)(xr + kb + 20));
    // A tile for K-dims 32..63
    pack4(a1, 0,  *(const float4*)(xr + 32 + kb));
    pack4(a1, 4,  *(const float4*)(xr + 32 + kb + 4));
    pack4(a1, 8,  *(const float4*)(xr + 32 + kb + 16));
    pack4(a1, 12, *(const float4*)(xr + 32 + kb + 20));

    v16h bm0 = *(const v16h*)(W1B + (size_t)(0 * 32 + lane) * 16);
    v16h bm1 = *(const v16h*)(W1B + (size_t)(1 * 32 + lane) * 16);

    v8f c = {};
    c = __builtin_amdgcn_wmma_f32_16x16x32_f16(false, a0, false, bm0, (short)0, c, false, false);
    c = __builtin_amdgcn_wmma_f32_16x16x32_f16(false, a1, false, bm1, (short)0, c, false, false);

    // C layout: VGPR v -> row M = v + hk*8, column N = s. Bias + ReLU + store f16.
    // Single exec region for the 8 stores (no per-store branches).
    bool full = (n0 + 16 <= Nv);
    if (s < R_CH) {
        float bias = b1[s];
        _Float16* dst = att_h + (size_t)(n0 + hk * 8) * R_CH + s;
        if (full) {
#pragma unroll
            for (int v = 0; v < 8; ++v)
                dst[(size_t)v * R_CH] = (_Float16)fmaxf(c[v] + bias, 0.0f);
        } else {
#pragma unroll
            for (int v = 0; v < 8; ++v)
                if (n0 + hk * 8 + v < Nv)
                    dst[(size_t)v * R_CH] = (_Float16)fmaxf(c[v] + bias, 0.0f);
        }
    }
}

// ---------------------------------------------------------------------------
// Fused: 4 dilated sparse convs (gather + masked WMMA einsum), cat @ W2 gate,
// out = x * relu(gate). One wave per 16-voxel tile; 7 WMMAs per branch.
// Gathers are unconditional (att is L2-resident) and masked via cndmask.
// ---------------------------------------------------------------------------
__global__ void __launch_bounds__(256)
lsa_gate(const float* __restrict__ x, const _Float16* __restrict__ att_h,
         const _Float16* __restrict__ WdB,
         const float* __restrict__ bd, const float* __restrict__ W2,
         const float* __restrict__ b2,
         const int* __restrict__ idx, const unsigned char* __restrict__ mask,
         float* __restrict__ out, int Nv) {
    int lane = threadIdx.x & 31;
    int wave = blockIdx.x * (blockDim.x >> 5) + (threadIdx.x >> 5);
    int tiles = (Nv + 15) >> 4;
    if (wave >= tiles) return;
    int n0 = wave << 4;
    int s  = lane & 15;
    int hk = lane >> 4;
    int nl = n0 + (lane & 15);           // A-matrix voxel row for gathers
    if (nl >= Nv) nl = Nv - 1;

    float b2v = b2[0];
    float w2a = (s < R_CH) ? W2[s] : 0.0f;

    // Gate partials: p[v] <-> voxel row v + hk*8 of this tile, channel s.
    // Branchless att contribution: address uses s&7 (valid), w2a==0 kills s>=8.
    float p[8];
    {
        const _Float16* ap = att_h + (size_t)(s & 7);
#pragma unroll
        for (int v = 0; v < 8; ++v) {
            int mr = n0 + hk * 8 + v;
            if (mr >= Nv) mr = Nv - 1;   // clamp; tail rows never stored
            p[v] = (float)ap[(size_t)mr * R_CH] * w2a;
        }
    }

    const v8h zero8 = {};
#pragma unroll
    for (int b = 0; b < B_BR; ++b) {
        const size_t jbase = (size_t)b * K_OFF * (size_t)Nv + nl;
        v8f c = {};
#pragma unroll
        for (int kblk = 0; kblk < KBLKS; ++kblk) {
            int k_lo = 4 * kblk + hk;            // <= 25, always valid
            int k_hi = k_lo + 2;                 // invalid only kblk==6,hk==1
            bool vhi = (k_hi < K_OFF);
            int k_hi_c = vhi ? k_hi : 0;         // clamp to a valid address
            size_t jlo = jbase + (size_t)k_lo * Nv;
            size_t jhi = jbase + (size_t)k_hi_c * Nv;
            // Streaming loads: idx/mask are read exactly once -> non-temporal.
            int tlo = __builtin_nontemporal_load(idx + jlo);
            int thi = __builtin_nontemporal_load(idx + jhi);
            bool mlo = __builtin_nontemporal_load(mask + jlo) != 0;
            bool mhi = vhi & (__builtin_nontemporal_load(mask + jhi) != 0);
            // Unconditional gathers (L2 hit), then select-zero (cndmask).
            v8h rlo = *(const v8h*)(att_h + (size_t)tlo * R_CH);
            v8h rhi = *(const v8h*)(att_h + (size_t)thi * R_CH);
            rlo = mlo ? rlo : zero8;
            rhi = mhi ? rhi : zero8;
            v16h a;
#pragma unroll
            for (int i = 0; i < 8; ++i) { a[i] = rlo[i]; a[8 + i] = rhi[i]; }
            v16h bm = *(const v16h*)(WdB + ((size_t)(b * KBLKS + kblk) * 32 + lane) * 16);
            // EXEC is all-1s here (no divergence in this loop body).
            c = __builtin_amdgcn_wmma_f32_16x16x32_f16(false, a, false, bm, (short)0, c, false, false);
        }
        float bds = (s < R_CH) ? bd[b * R_CH + s] : 0.0f;
        float w2d = (s < R_CH) ? W2[R_CH + b * R_CH + s] : 0.0f;
#pragma unroll
        for (int v = 0; v < 8; ++v) {
            float dv = fmaxf(c[v] + bds, 0.0f);    // relu(d)
            p[v] = fmaf(dv, w2d, p[v]);
        }
    }

    // Reduce over the 8 channel lanes (xor 1,2,4 stays inside each 8-lane group;
    // lanes 8..15 / 24..31 carry zeros). Then gate = relu(. + b2).
#pragma unroll
    for (int v = 0; v < 8; ++v) {
        float t = p[v];
        t += __shfl_xor(t, 1, 32);
        t += __shfl_xor(t, 2, 32);
        t += __shfl_xor(t, 4, 32);
        p[v] = fmaxf(t + b2v, 0.0f);
    }

    // out[n, :] = x[n, :] * gate[n]; 64 floats per row, v2f per lane.
    // Uniform full-tile test keeps the hot path free of per-lane guards;
    // output is never re-read -> non-temporal stores.
    bool full = (n0 + 16 <= Nv);
    if (full) {
#pragma unroll
        for (int mm = 0; mm < 16; ++mm) {
            float gm = __shfl(p[mm & 7], (mm < 8) ? 0 : 16, 32);
            const v2f* xr2 = (const v2f*)(x + (size_t)(n0 + mm) * C_CH);
            v2f v2 = xr2[lane];
            v2 *= gm;
            __builtin_nontemporal_store(v2, (v2f*)(out + (size_t)(n0 + mm) * C_CH) + lane);
        }
    } else {
#pragma unroll
        for (int mm = 0; mm < 16; ++mm) {
            int n = n0 + mm;
            if (n < Nv) {
                float gm = __shfl(p[mm & 7], (mm < 8) ? 0 : 16, 32);
                const v2f* xr2 = (const v2f*)(x + (size_t)n * C_CH);
                v2f v2 = xr2[lane];
                v2 *= gm;
                __builtin_nontemporal_store(v2, (v2f*)(out + (size_t)n * C_CH) + lane);
            }
        }
    }
}

extern "C" void kernel_launch(void* const* d_in, const int* in_sizes, int n_in,
                              void* d_out, int out_size, void* d_ws, size_t ws_size,
                              hipStream_t stream) {
    const float* x  = (const float*)d_in[0];
    const float* W1 = (const float*)d_in[1];
    const float* b1 = (const float*)d_in[2];
    const float* Wd = (const float*)d_in[3];
    const float* bd = (const float*)d_in[4];
    const float* W2 = (const float*)d_in[5];
    const float* b2 = (const float*)d_in[6];
    const int*   idx = (const int*)d_in[7];
    const unsigned char* mask = (const unsigned char*)d_in[8];   // jnp bool -> 1 byte
    float* out = (float*)d_out;

    int Nv = in_sizes[0] / C_CH;

    // Workspace: att_h (N*8 f16) | WdB (4*7*32*16 f16) | W1B (2*32*16 f16)
    char* ws = (char*)d_ws;
    size_t att_bytes = (((size_t)Nv * R_CH * sizeof(_Float16)) + 255) & ~(size_t)255;
    _Float16* att_h = (_Float16*)ws;
    _Float16* WdB   = (_Float16*)(ws + att_bytes);
    _Float16* W1B   = (_Float16*)(ws + att_bytes + (size_t)B_BR * KBLKS * 32 * 16 * sizeof(_Float16));

    lsa_prep<<<4, 256, 0, stream>>>(W1, Wd, W1B, WdB);

    int tiles = (Nv + 15) / 16;          // 9375 for N=150000
    int blocks = (tiles + 7) / 8;        // 8 waves (256 threads) per block
    lsa_att<<<blocks, 256, 0, stream>>>(x, b1, W1B, att_h, Nv);
    lsa_gate<<<blocks, 256, 0, stream>>>(x, att_h, WdB, bd, W2, b2, idx, mask, out, Nv);
}
